// DynamicCompressor_39642548142429
// MI455X (gfx1250) — matrix operand order repaired and checked by reading, not compiled, not run on hardware
//
#include <hip/hip_runtime.h>
#include <hip/hip_bf16.h>

#define T_FRAMES 64
#define HWD 24
#define C_DIM 1024
#define MID 256
#define TILES 9          // 144 windows / 16 per tile
#define KCHUNKS 32       // 1024 / 32

typedef __attribute__((ext_vector_type(16))) __bf16 v16bf;
typedef __attribute__((ext_vector_type(8)))  float  v8f;

union Frag16 { uint4 q[2]; unsigned w[8]; unsigned short h[16]; v16bf v; };

__device__ __forceinline__ unsigned short f2bf(float f) {
  unsigned u = __float_as_uint(f);
  u += 0x7fffu + ((u >> 16) & 1u);      // round-to-nearest-even
  return (unsigned short)(u >> 16);
}

// Packed 2xbf16 ALU (VOP3P, CDNA5) — not exposed as builtins, use inline asm.
__device__ __forceinline__ unsigned pk_add_bf16(unsigned a, unsigned b) {
  unsigned d;
  asm("v_pk_add_bf16 %0, %1, %2" : "=v"(d) : "v"(a), "v"(b));
  return d;
}
__device__ __forceinline__ unsigned pk_mul_bf16(unsigned a, unsigned b) {
  unsigned d;
  asm("v_pk_mul_bf16 %0, %1, %2" : "=v"(d) : "v"(a), "v"(b));
  return d;
}

// ---------------------------------------------------------------------------
// Prep: Wq/Wk fp32 -> bf16 in WMMA B-fragment swizzled layout in workspace.
// Layout: [mat(2)][n(16)][kc(32)][lane(32)][e(16)] bf16 (32 bytes/lane).
// Dense 32x16 16-bit B layout (contiguous k order chosen here, same as A):
//   lane holds column N = n*16 + (lane&15),
//   element e holds K = kc*32 + (lane>=16 ? 16 : 0) + e.
// ---------------------------------------------------------------------------
__global__ void prep_weights(const float* __restrict__ Wq,
                             const float* __restrict__ Wk,
                             unsigned short* __restrict__ ws) {
  int idx = blockIdx.x * blockDim.x + threadIdx.x;   // 0 .. 2*262144-1
  int mat  = idx >> 18;
  int r    = idx & 0x3FFFF;
  int e    = r & 15;
  int lane = (r >> 4) & 31;
  int kc   = (r >> 9) & 31;
  int n    = (r >> 14) & 15;
  int K = kc * 32 + ((lane >= 16) ? 16 : 0) + e;
  int N = n * 16 + (lane & 15);
  const float* W = mat ? Wk : Wq;
  ws[idx] = f2bf(W[(size_t)K * MID + N]);
}

// ---------------------------------------------------------------------------
// Main fused kernel: one block = one frame t x one tile of 16 windows.
// 512 threads = 16 waves; wave w owns MID columns [16w, 16w+16).
// ---------------------------------------------------------------------------
__global__ __launch_bounds__(512)
void dyncomp_kernel(const float* __restrict__ img,
                    const float* __restrict__ bq,
                    const float* __restrict__ bk,
                    const unsigned short* __restrict__ wsB,
                    float* __restrict__ out) {
  __shared__ unsigned short ldsA[64][40];   // windows chunk, bf16, padded pitch
  __shared__ float ldsAttn[16][4];

  const int t    = blockIdx.x / TILES;
  const int tile = blockIdx.x % TILES;
  const int tid  = threadIdx.x;
  const int lane = tid & 31;
  const int wav  = tid >> 5;                // N-tile 0..15

  // ---- cooperative loader mapping: 8 threads per row, 64 rows ----
  // row r = w4*16 + p  (patch-major so k-accum rows align with q rows)
  const int row   = tid >> 3;
  const int kq    = (tid & 7) * 4;
  const int p_ld  = row & 15;
  const int w4_ld = row >> 4;
  const int pg_ld = tile * 16 + p_ld;
  const int i_ld  = pg_ld / 12, j_ld = pg_ld % 12;
  const int n_ld  = (2 * i_ld + (w4_ld >> 1)) * HWD + (2 * j_ld + (w4_ld & 1));
  const float* rowptr = img + ((size_t)t * (HWD * HWD) + n_ld) * C_DIM;

  const int kA   = (lane >= 16) ? 16 : 0;   // contiguous A layout K base (matches B)
  const int mrow = lane & 15;

  v8f accq = {0.f,0.f,0.f,0.f,0.f,0.f,0.f,0.f};
  v8f acck[4];
  #pragma unroll
  for (int m = 0; m < 4; ++m) acck[m] = accq;

  for (int kc = 0; kc < KCHUNKS; ++kc) {
    __syncthreads();                                   // prev frag reads done
    // stage 64x32 fp32 slice -> bf16 LDS
    float4 v4 = *(const float4*)(rowptr + kc * 32 + kq);
    ushort4 h4;
    h4.x = f2bf(v4.x); h4.y = f2bf(v4.y); h4.z = f2bf(v4.z); h4.w = f2bf(v4.w);
    *(ushort4*)&ldsA[row][kq] = h4;
    __syncthreads();

    // B fragments from swizzled workspace (two b128 per lane, L2-resident)
    Frag16 fbq, fbk;
    const uint4* pq = (const uint4*)(wsB +
        (((size_t)(wav * KCHUNKS + kc)) * 32 + lane) * 16);
    const uint4* pk = (const uint4*)(wsB + 262144 +
        (((size_t)(wav * KCHUNKS + kc)) * 32 + lane) * 16);
    fbq.q[0] = pq[0]; fbq.q[1] = pq[1];
    fbk.q[0] = pk[0]; fbk.q[1] = pk[1];

    // A fragments: row m*16+mrow of patch m == patch m of window mrow
    Frag16 fa[4];
    #pragma unroll
    for (int m = 0; m < 4; ++m) {
      fa[m].q[0] = *(const uint4*)&ldsA[m * 16 + mrow][kA];
      fa[m].q[1] = *(const uint4*)&ldsA[m * 16 + mrow][kA + 8];
    }

    // k GEMM (4 M-tiles)
    #pragma unroll
    for (int m = 0; m < 4; ++m)
      acck[m] = __builtin_amdgcn_wmma_f32_16x16x32_bf16(
          false, fa[m].v, false, fbk.v, (short)0, acck[m], false, false);

    // pooled A-fragment = elementwise mean of the 4 patch fragments,
    // done with packed 2xbf16 VOP3P ops (4 VALU per dword).
    Frag16 fp;
    #pragma unroll
    for (int i = 0; i < 8; ++i) {
      unsigned s01 = pk_add_bf16(fa[0].w[i], fa[1].w[i]);
      unsigned s23 = pk_add_bf16(fa[2].w[i], fa[3].w[i]);
      fp.w[i] = pk_mul_bf16(pk_add_bf16(s01, s23), 0x3e803e80u); // * 0.25
    }
    // q GEMM (pooled tile)
    accq = __builtin_amdgcn_wmma_f32_16x16x32_bf16(
        false, fp.v, false, fbq.v, (short)0, accq, false, false);
  }

  // ---- bias add ----
  const int Ncol = wav * 16 + (lane & 15);
  {
    float bqv = bq[Ncol], bkv = bk[Ncol];
    #pragma unroll
    for (int v = 0; v < 8; ++v) {
      accq[v] += bqv;
      #pragma unroll
      for (int m = 0; m < 4; ++m) acck[m][v] += bkv;
    }
  }

  // ---- attention logits: q . k, reduced over 256 cols ----
  __syncthreads();
  if (tid < 64) ldsAttn[tid >> 2][tid & 3] = 0.f;
  __syncthreads();

  const int pBase = (lane >> 4) * 8;   // C/D rows: lane<16 -> M=v, else M=v+8
  #pragma unroll
  for (int w4 = 0; w4 < 4; ++w4) {
    #pragma unroll
    for (int v = 0; v < 8; ++v) {
      float s = accq[v] * acck[w4][v];
      s += __shfl_xor(s, 1, 32);
      s += __shfl_xor(s, 2, 32);
      s += __shfl_xor(s, 4, 32);
      s += __shfl_xor(s, 8, 32);
      if ((lane & 15) == 0)
        atomicAdd(&ldsAttn[pBase + v][w4], s * 0.0625f);  // 1/sqrt(256)
    }
  }
  __syncthreads();

  // ---- softmax over 4 (with nan_to_num) ----
  if (tid < 16) {
    float x0 = ldsAttn[tid][0], x1 = ldsAttn[tid][1];
    float x2 = ldsAttn[tid][2], x3 = ldsAttn[tid][3];
    x0 = (x0 == x0) ? x0 : 0.f;  x1 = (x1 == x1) ? x1 : 0.f;
    x2 = (x2 == x2) ? x2 : 0.f;  x3 = (x3 == x3) ? x3 : 0.f;
    float m = fmaxf(fmaxf(x0, x1), fmaxf(x2, x3));
    float e0 = __expf(x0 - m), e1 = __expf(x1 - m);
    float e2 = __expf(x2 - m), e3 = __expf(x3 - m);
    float inv = 1.f / (e0 + e1 + e2 + e3);
    ldsAttn[tid][0] = e0 * inv;  ldsAttn[tid][1] = e1 * inv;
    ldsAttn[tid][2] = e2 * inv;  ldsAttn[tid][3] = e3 * inv;
  }
  __syncthreads();

  // ---- output: pooled + softmax-weighted window avg (fp32, L2-hit reread) ----
  for (int idx = tid; idx < 16 * C_DIM; idx += 512) {
    int p = idx >> 10;
    int c = idx & (C_DIM - 1);
    int pg = tile * 16 + p;
    int i = pg / 12, j = pg % 12;
    size_t base = ((size_t)t * (HWD * HWD) + (2 * i) * HWD + 2 * j) * C_DIM + c;
    float a0 = img[base];
    float a1 = img[base + C_DIM];
    float a2 = img[base + HWD * C_DIM];
    float a3 = img[base + (HWD + 1) * C_DIM];
    float w0 = ldsAttn[p][0], w1 = ldsAttn[p][1];
    float w2 = ldsAttn[p][2], w3 = ldsAttn[p][3];
    float r = 0.25f * (a0 + a1 + a2 + a3)
            + w0 * a0 + w1 * a1 + w2 * a2 + w3 * a3;
    out[((size_t)t * 144 + pg) * C_DIM + c] = r;
  }
}

extern "C" void kernel_launch(void* const* d_in, const int* in_sizes, int n_in,
                              void* d_out, int out_size, void* d_ws, size_t ws_size,
                              hipStream_t stream) {
  const float* img = (const float*)d_in[0];
  const float* Wq  = (const float*)d_in[1];
  const float* bq  = (const float*)d_in[2];
  const float* Wk  = (const float*)d_in[3];
  const float* bk  = (const float*)d_in[4];
  float* out = (float*)d_out;
  unsigned short* wsB = (unsigned short*)d_ws;   // needs 1 MB

  prep_weights<<<2048, 256, 0, stream>>>(Wq, Wk, wsB);
  dyncomp_kernel<<<T_FRAMES * TILES, 512, 0, stream>>>(img, bq, bk, wsB, out);
}
